// FBAMSparseMemoryAgent_14723147891394
// MI455X (gfx1250) — matrix operand order repaired and checked by reading, not compiled, not run on hardware
//
#include <hip/hip_runtime.h>
#include <hip/hip_bf16.h>

// ---------------------------------------------------------------------------
// FBAM sparse-memory retrieval for MI455X (gfx1250, wave32, WMMA).
//
// score[b,m] = 2*(q_b . mem_m) - |mem_m|^2  (== -dist + const(b); identical
//                                            top-k and softmax weights)
// q = h @ Wq + bq in f16 WMMA (f32 accumulate). Fused kernel: 32 query rows
// per block (2 A-tiles x 8 waves) halves L2 traffic vs 16-row blocks while
// keeping 256 blocks to fill the WGPs; M is processed in 2 chunks of 2048
// with a 256 KB LDS score panel (CDNA5: 320 KB/WGP) and a carried top-16.
// ---------------------------------------------------------------------------

#define B_  8192
#define H_  512
#define M_  4096
#define D_  256
#define K_  16

#define ROWS_PB   32          // query rows per block
#define CHUNK_    2048        // M columns per LDS chunk
#define NCHUNK_   (M_ / CHUNK_)
#define RUN_FLAG  0x40000000  // candidate came from the carried top list

typedef __attribute__((ext_vector_type(16))) _Float16 v16h;
typedef __attribute__((ext_vector_type(8)))  _Float16 v8h;
typedef __attribute__((ext_vector_type(8)))  float    v8f;

static __device__ __forceinline__ v16h join16(v8h lo, v8h hi) {
  return __builtin_shufflevector(lo, hi, 0,1,2,3,4,5,6,7,8,9,10,11,12,13,14,15);
}

// ---------------- prep kernels -------------------------------------------

__global__ __launch_bounds__(256) void conv_f16_kernel(const float* __restrict__ src,
                                                       _Float16* __restrict__ dst, int n) {
  int i = blockIdx.x * 256 + threadIdx.x;
  if (i < n) dst[i] = (_Float16)src[i];
}

// Wq [H,D] row-major -> wqT [D,H] f16 (K-contiguous per output column)
__global__ __launch_bounds__(256) void transpose_wq_kernel(const float* __restrict__ Wq,
                                                           _Float16* __restrict__ wqT) {
  int t = blockIdx.x * 256 + threadIdx.x;   // H_*D_ threads
  int h = t >> 8;                           // / D_
  int d = t & (D_ - 1);
  wqT[d * H_ + h] = (_Float16)Wq[t];
}

// mem [M,D] -> f16 copy + per-row sum of squares
__global__ __launch_bounds__(256) void prep_mem_kernel(const float* __restrict__ mem,
                                                       _Float16* __restrict__ memf16,
                                                       float* __restrict__ msq) {
  __shared__ float red[256];
  int m = blockIdx.x, d = threadIdx.x;
  float x = mem[m * D_ + d];
  memf16[m * D_ + d] = (_Float16)x;
  red[d] = x * x;
  __syncthreads();
  for (int s = 128; s > 0; s >>= 1) {
    if (d < s) red[d] += red[d + s];
    __syncthreads();
  }
  if (d == 0) msq[m] = red[0];
}

// ---------------- q = h @ Wq + bq  (f16 WMMA, one 16x16 tile per wave) ----

__global__ __launch_bounds__(256) void proj_kernel(const _Float16* __restrict__ hf16,
                                                   const _Float16* __restrict__ wqT,
                                                   const float* __restrict__ bq,
                                                   _Float16* __restrict__ qf16) {
  int wave = threadIdx.x >> 5, lane = threadIdx.x & 31;
  int tile = blockIdx.x * 8 + wave;        // (B_/16)*(D_/16) = 8192 tiles
  int row0 = (tile >> 4) * 16;             // D_/16 = 16 col-tiles
  int col0 = (tile & 15) * 16;
  int l16 = lane & 15, hi = lane >> 4;
  int arow = row0 + l16;                   // A: lane L holds row M = L%16

  v8f c = {};
#pragma unroll
  for (int kc = 0; kc < H_ / 32; ++kc) {
    int K0 = kc * 32;
    // A 16x32 f16: lanes<16 hold K 0-7 & 16-23, lanes>=16 hold K 8-15 & 24-31
    v8h alo = *(const v8h*)(hf16 + (size_t)arow * H_ + K0 + hi * 8);
    v8h ahi = *(const v8h*)(hf16 + (size_t)arow * H_ + K0 + 16 + hi * 8);
    v16h a = join16(alo, ahi);
    // B 32x16 f16: lane holds column N=l16; lanes<16 K 0-7 & 16-23, lanes>=16 K 8-15 & 24-31
    v16h b = join16(*(const v8h*)(wqT + (size_t)(col0 + l16) * H_ + K0 + hi * 8),
                    *(const v8h*)(wqT + (size_t)(col0 + l16) * H_ + K0 + 16 + hi * 8));
    c = __builtin_amdgcn_wmma_f32_16x16x32_f16(false, a, false, b, (short)0, c,
                                               false, false);
  }
  float bias = bq[col0 + l16];
  // C/D layout: lane holds col N=l16; VGPR r -> row (hi*8 + r)
#pragma unroll
  for (int r = 0; r < 8; ++r) {
    int orow = row0 + hi * 8 + r;
    qf16[(size_t)orow * D_ + col0 + l16] = (_Float16)(c[r] + bias);
  }
}

// ---------------- fused: scores (WMMA) + chunked top-16 + softmax + gather

__global__ __launch_bounds__(256) void fused_kernel(const _Float16* __restrict__ qf16,
                                                    const _Float16* __restrict__ memf16,
                                                    const float* __restrict__ msq,
                                                    const float* __restrict__ memf32,
                                                    float* __restrict__ out) {
  extern __shared__ char smem[];
  float* s_sc   = (float*)smem;                                   // ROWS_PB*CHUNK_
  float* s_runv = (float*)(smem + (size_t)ROWS_PB * CHUNK_ * 4);  // ROWS_PB*16
  int*   s_runi = (int*)  (smem + (size_t)ROWS_PB * CHUNK_ * 4 + ROWS_PB * K_ * 4);

  int wave = threadIdx.x >> 5, lane = threadIdx.x & 31;
  int l16 = lane & 15, hi = lane >> 4;
  int rt = wave & 1;            // which 16-row A-tile this wave owns
  int quarter = wave >> 1;      // which quarter of the chunk's columns
  int row0 = blockIdx.x * ROWS_PB;

  // init carried top lists (wave w owns rows w*4 .. w*4+3 in merge phase)
  if (lane < K_) {
#pragma unroll
    for (int i = 0; i < 4; ++i) {
      s_runv[(wave * 4 + i) * K_ + lane] = -__builtin_inff();
      s_runi[(wave * 4 + i) * K_ + lane] = 0;
    }
  }

  // cache this wave's 16x256 A-tile (q rows) in registers: 8 K-chunks
  v16h aReg[8];
  {
    int arow = row0 + rt * 16 + l16;
#pragma unroll
    for (int kc = 0; kc < 8; ++kc) {
      v8h alo = *(const v8h*)(qf16 + (size_t)arow * D_ + kc * 32 + hi * 8);
      v8h ahi = *(const v8h*)(qf16 + (size_t)arow * D_ + kc * 32 + 16 + hi * 8);
      aReg[kc] = join16(alo, ahi);
    }
  }

  for (int chunk = 0; chunk < NCHUNK_; ++chunk) {
    // --- GEMM: pairs of waves (rt=0/1) share the same B-tile sequence ----
    for (int t = 0; t < 32; ++t) {
      int mt = chunk * (CHUNK_ / 16) + quarter * 32 + t;
      int n = mt * 16 + l16;               // this lane's memory-slot column
      if (t + 1 < 32)
        __builtin_prefetch(memf16 + (size_t)(n + 16) * D_, 0, 0);
      v8f c = {};
#pragma unroll
      for (int kc = 0; kc < 8; ++kc) {
        v16h b = join16(*(const v8h*)(memf16 + (size_t)n * D_ + kc * 32 + hi * 8),
                        *(const v8h*)(memf16 + (size_t)n * D_ + kc * 32 + 16 + hi * 8));
        c = __builtin_amdgcn_wmma_f32_16x16x32_f16(false, aReg[kc], false, b,
                                                   (short)0, c, false, false);
      }
      float mq = msq[n];
      int cc = (quarter * 32 + t) * 16 + l16;   // chunk-local column
#pragma unroll
      for (int r = 0; r < 8; ++r)
        s_sc[(size_t)(rt * 16 + hi * 8 + r) * CHUNK_ + cc] = 2.0f * c[r] - mq;
    }
    __syncthreads();

    // --- merge chunk into carried top-16 (4 rows per wave) ---------------
    for (int i = 0; i < 4; ++i) {
      int lrow = wave * 4 + i;
      float* srow = s_sc + (size_t)lrow * CHUNK_;
      float newv = -__builtin_inff();
      int   newi = 0;
      for (int it = 0; it < K_; ++it) {
        float best = -__builtin_inff();
        int enc = lane;
        for (int j = lane; j < CHUNK_; j += 32) {
          float x = srow[j];
          if (x > best || (x == best && j < enc)) { best = x; enc = j; }
        }
        if (lane < K_) {                       // lane k champions carry-entry k
          float rv = s_runv[lrow * K_ + lane];
          int re = RUN_FLAG | lane;
          if (rv > best) { best = rv; enc = re; }
        }
#pragma unroll
        for (int off = 16; off >= 1; off >>= 1) {   // argmax: (val desc, enc asc)
          float ov = __shfl_xor(best, off, 32);
          int   oe = __shfl_xor(enc, off, 32);
          if (ov > best || (ov == best && oe < enc)) { best = ov; enc = oe; }
        }
        if (enc & RUN_FLAG) {
          int k2 = enc & 15;
          if (lane == it) { newv = best; newi = s_runi[lrow * K_ + k2]; }
          if (lane == k2) s_runv[lrow * K_ + k2] = -__builtin_inff(); // same-lane RAW
        } else {
          if (lane == it) { newv = best; newi = chunk * CHUNK_ + enc; }
          if ((enc & 31) == lane) srow[enc] = -__builtin_inff();      // same-lane RAW
        }
      }
      if (lane < K_) {                         // winner #it lives in lane it
        s_runv[lrow * K_ + lane] = newv;
        s_runi[lrow * K_ + lane] = newi;
      }
    }
    __syncthreads();                           // panel reused by next chunk
  }

  // --- softmax over final top-16 + f32 gather (4 rows per wave) ----------
  for (int i = 0; i < 4; ++i) {
    int lrow = wave * 4 + i;
    float v  = (lane < K_) ? s_runv[lrow * K_ + lane] : -__builtin_inff();
    float mx = __shfl(v, 0, 32);               // list is in descending order
    float e  = (lane < K_) ? __expf(v - mx) : 0.0f;
    float s  = e;
#pragma unroll
    for (int off = 16; off >= 1; off >>= 1) s += __shfl_xor(s, off, 32);
    float wn = e / s;

    float acc[8];
#pragma unroll
    for (int j = 0; j < 8; ++j) acc[j] = 0.0f;
#pragma unroll
    for (int k = 0; k < K_; ++k) {
      float wk = __shfl(wn, k, 32);
      int mi = s_runi[lrow * K_ + k];
      const float* mrow = memf32 + (size_t)mi * D_;   // 4 MB table: L2-resident
#pragma unroll
      for (int j = 0; j < 8; ++j) acc[j] += wk * mrow[lane + 32 * j];
    }
    float* orow = out + (size_t)(row0 + lrow) * D_;
#pragma unroll
    for (int j = 0; j < 8; ++j) orow[lane + 32 * j] = acc[j];
  }
}

// ---------------- host launcher ------------------------------------------

extern "C" void kernel_launch(void* const* d_in, const int* in_sizes, int n_in,
                              void* d_out, int out_size, void* d_ws, size_t ws_size,
                              hipStream_t stream) {
  const float* h   = (const float*)d_in[0];   // [B,H]
  const float* mem = (const float*)d_in[1];   // [M,D]
  const float* Wq  = (const float*)d_in[2];   // [H,D]
  const float* bq  = (const float*)d_in[3];   // [D]
  // d_in[4] = k (==16, baked in at compile time)
  float* out = (float*)d_out;

  char* w = (char*)d_ws;
  const size_t MB = 1024 * 1024;
  _Float16* hf16   = (_Float16*)(w);                               // 8 MB
  _Float16* wqT    = (_Float16*)(w + 8 * MB);                      // 256 KB
  _Float16* memf16 = (_Float16*)(w + 8 * MB + 256 * 1024);         // 2 MB
  float*    msq    = (float*)   (w + 10 * MB + 256 * 1024);        // 16 KB
  _Float16* qf16   = (_Float16*)(w + 10 * MB + 320 * 1024);        // 4 MB

  conv_f16_kernel<<<(B_ * H_) / 256, 256, 0, stream>>>(h, hf16, B_ * H_);
  transpose_wq_kernel<<<(H_ * D_) / 256, 256, 0, stream>>>(Wq, wqT);
  prep_mem_kernel<<<M_, 256, 0, stream>>>(mem, memf16, msq);

  proj_kernel<<<(B_ / 16) * (D_ / 16) / 8, 256, 0, stream>>>(hf16, wqT, bq, qf16);

  // 32*2048*4 + 32*16*4 + 32*16*4 = 266,240 B  (< 320 KB/WGP on CDNA5)
  size_t smem = (size_t)ROWS_PB * CHUNK_ * 4 + (size_t)ROWS_PB * K_ * 4 * 2;
  fused_kernel<<<B_ / ROWS_PB, 256, smem, stream>>>(qf16, memf16, msq, mem, out);
}
